// SoftDDT_88124138979497
// MI455X (gfx1250) — matrix operand order repaired and checked by reading, compiled, not run on hardware
//
#include <hip/hip_runtime.h>
#include <hip/hip_bf16.h>

// ---------------------------------------------------------------------------
// Soft decision tree, MAX_DEPTH=12, B=4096, IN=1024, OUT=1024.
//   g     = sigmoid(x @ W^T + b)        : 4096 x 4095  (GEMM1, fused epilogue)
//   probs = tree-expand(g)              : 4096 x 4096
//   out   = probs @ leaf                : 4096 x 1024  (GEMM2)
// GEMMs: v_wmma_f32_16x16x32_f16, LDS staged via GLOBAL_LOAD_ASYNC_TO_LDS_B128
// (ASYNCcnt-tracked, double buffered, one barrier per K-step).
// Workspace layout (88 MiB):
//   [0      , 8Mi )  x16    : 4096x1024 f16
//   [8Mi    , 16Mi)  W16    : 4096x1024 f16 (row 4095 zero-padded)
//   [16Mi   , 24Mi)  leafT16: 1024x4096 f16 (transposed: N x K)
//   [24Mi   , 56Mi)  g16    : 4096x4096 f16
//   [56Mi   , 88Mi)  p16    : 4096x4096 f16
// ---------------------------------------------------------------------------

typedef _Float16 v16h __attribute__((ext_vector_type(16)));
typedef _Float16 v8h  __attribute__((ext_vector_type(8)));
typedef _Float16 v4h  __attribute__((ext_vector_type(4)));
typedef float    v8f  __attribute__((ext_vector_type(8)));

#define LDH 40   // LDS tile row stride in halves (32 data + 8 pad: conflict-free b128)

// ----- CDNA5 async memory->LDS copy (16B per lane), GVS addressing ---------
template <int OFS>
__device__ __forceinline__ void async_b128(uint32_t lds_byte_addr,
                                           uint32_t glb_byte_off,
                                           const void* sbase) {
  asm volatile("global_load_async_to_lds_b128 %0, %1, %2 offset:%3"
               :: "v"(lds_byte_addr), "v"(glb_byte_off), "s"(sbase), "n"(OFS)
               : "memory");
}
__device__ __forceinline__ void wait_asynccnt0() {
  asm volatile("s_wait_asynccnt 0" ::: "memory");
}
__device__ __forceinline__ uint32_t lds_addr(const void* p) {
  return (uint32_t)(uintptr_t)p;   // LDS aperture: low 32 bits = LDS byte offset
}

// ----- f32 -> f16 convert (4 elements / thread), zero-fills the padded tail -----
__global__ __launch_bounds__(256) void cvt_f16_kernel(
    const float* __restrict__ src, _Float16* __restrict__ dst,
    int n_valid4, int n_total4) {
  int i = blockIdx.x * 256 + threadIdx.x;
  if (i >= n_total4) return;
  float4 v = make_float4(0.f, 0.f, 0.f, 0.f);
  if (i < n_valid4) v = ((const float4*)src)[i];
  v4h h;
  h[0] = (_Float16)v.x; h[1] = (_Float16)v.y;
  h[2] = (_Float16)v.z; h[3] = (_Float16)v.w;
  *(v4h*)(dst + (size_t)i * 4) = h;
}

// ----- leaf (4096 x 1024 f32) -> leafT16 (1024 x 4096 f16), LDS-tiled -----
__global__ __launch_bounds__(256) void transpose_cvt_kernel(
    const float* __restrict__ src, _Float16* __restrict__ dst) {
  __shared__ float tile[32][33];
  int o0 = blockIdx.x * 32;       // output-feature dim (1024)
  int l0 = blockIdx.y * 32;       // leaf dim (4096)
  int tx = threadIdx.x;           // 0..31
  int ty = threadIdx.y;           // 0..7
#pragma unroll
  for (int j = 0; j < 32; j += 8)
    tile[ty + j][tx] = src[(size_t)(l0 + ty + j) * 1024 + (o0 + tx)];
  __syncthreads();
#pragma unroll
  for (int j = 0; j < 32; j += 8)
    dst[(size_t)(o0 + ty + j) * 4096 + (l0 + tx)] = (_Float16)tile[tx][ty + j];
}

// ----- WMMA fragment gathers from LDS (wave32 layouts per CDNA5 ISA 7.12.2) -----
// A 16x32 f16: VGPR 0-3 : lanes 0-15 K=0..7,  lanes 16-31 K=8..15
//              VGPR 4-7 : lanes 0-15 K=16..23, lanes 16-31 K=24..31
__device__ inline v16h load_frag_a(const _Float16* base, int lane) {
  int row  = lane & 15;
  int kofs = (lane >> 4) << 3;              // 0 or 8
  const _Float16* p = base + row * LDH + kofs;
  v8h lo = *(const v8h*)(p);                // K kofs .. kofs+7
  v8h hi = *(const v8h*)(p + 16);           // K 16+kofs .. 16+kofs+7
  return __builtin_shufflevector(lo, hi, 0,1,2,3,4,5,6,7,8,9,10,11,12,13,14,15);
}
// B 32x16 f16 (stored here as N x K rows): same K order as A (lanes 0-15 K=0..7,16..23; lanes 16-31 K=8..15,24..31)
__device__ inline v16h load_frag_b(const _Float16* base, int lane) {
  int row  = lane & 15;
  int kofs = (lane >> 4) << 3;              // 0 or 8
  const _Float16* p = base + row * LDH + kofs;
  v8h lo = *(const v8h*)(p);
  v8h hi = *(const v8h*)(p + 16);
  return __builtin_shufflevector(lo, hi, 0,1,2,3,4,5,6,7,8,9,10,11,12,13,14,15);
}

// ----- Tiled WMMA GEMM: C(MxN) = A(MxK,f16,row-major) * B(NxK,f16,row-major)^T -----
// Block tile 128x128, 256 threads = 8 waves, wave tile 32x64 (2x4 16x16 accs).
// FUSE_SIGMOID: Cout is f16, epilogue = sigmoid(acc + bias[n]); else Cout f32.
template <bool FUSE_SIGMOID>
__global__ __launch_bounds__(256) void wmma_gemm_kernel(
    const _Float16* __restrict__ A, const _Float16* __restrict__ B,
    const float* __restrict__ bias, void* __restrict__ Cout,
    int K, int ldc, int nbias) {
  __shared__ _Float16 lsA[2][128 * LDH];
  __shared__ _Float16 lsB[2][128 * LDH];

  const int tid  = threadIdx.x;
  const int lane = tid & 31;
  const int wave = tid >> 5;
  const int wm = wave & 3;                  // wave row   (4 x 32 = 128)
  const int wn = wave >> 2;                 // wave col   (2 x 64 = 128)
  const int blockM = blockIdx.y * 128;
  const int blockN = blockIdx.x * 128;

  // Staging: each thread owns one 32-byte chunk (two b128) per matrix per step.
  const int lrow = tid >> 1;                // 0..127
  const int lcol = (tid & 1) << 4;          // 0 or 16 halves

  const uint32_t lA0 = lds_addr(&lsA[0][lrow * LDH + lcol]);
  const uint32_t lA1 = lds_addr(&lsA[1][lrow * LDH + lcol]);
  const uint32_t lB0 = lds_addr(&lsB[0][lrow * LDH + lcol]);
  const uint32_t lB1 = lds_addr(&lsB[1][lrow * LDH + lcol]);
  const uint32_t vA = (uint32_t)(((blockM + lrow) * K + lcol) * 2);  // bytes
  const uint32_t vB = (uint32_t)(((blockN + lrow) * K + lcol) * 2);  // bytes

  v8f acc[2][4];
  const v8f vzero = {0.f, 0.f, 0.f, 0.f, 0.f, 0.f, 0.f, 0.f};
#pragma unroll
  for (int mi = 0; mi < 2; ++mi)
#pragma unroll
    for (int ni = 0; ni < 4; ++ni) acc[mi][ni] = vzero;

  // Prologue: async-stage K-step 0 into buffer 0.
  async_b128<0>(lA0, vA, A);
  async_b128<16>(lA0, vA, A);
  async_b128<0>(lB0, vB, B);
  async_b128<16>(lB0, vB, B);

  const int nk = K >> 5;                    // 32-wide K steps
  int buf = 0;
  for (int kt = 0; kt < nk; ++kt) {
    wait_asynccnt0();                       // own async copies landed in LDS
    __syncthreads();                        // all waves' copies landed; prior reads drained

    if (kt + 1 < nk) {                      // stage next tile into the other buffer
      const uint32_t kb = (uint32_t)(kt + 1) * 64;
      const uint32_t la = buf ? lA0 : lA1;
      const uint32_t lb = buf ? lB0 : lB1;
      async_b128<0>(la, vA + kb, A);
      async_b128<16>(la, vA + kb, A);
      async_b128<0>(lb, vB + kb, B);
      async_b128<16>(lb, vB + kb, B);
    }

    v16h aF[2], bF[4];
#pragma unroll
    for (int mi = 0; mi < 2; ++mi)
      aF[mi] = load_frag_a(&lsA[buf][(wm * 32 + mi * 16) * LDH], lane);
#pragma unroll
    for (int ni = 0; ni < 4; ++ni)
      bF[ni] = load_frag_b(&lsB[buf][(wn * 64 + ni * 16) * LDH], lane);
#pragma unroll
    for (int mi = 0; mi < 2; ++mi)
#pragma unroll
      for (int ni = 0; ni < 4; ++ni)
        acc[mi][ni] = __builtin_amdgcn_wmma_f32_16x16x32_f16(
            false, aF[mi], false, bF[ni], (short)0, acc[mi][ni], false, false);
    buf ^= 1;
  }

  // Epilogue. C/D layout: VGPR v -> row v (lanes 0-15) / v+8 (lanes 16-31), col = lane&15.
  const int colBase = blockN + wn * 64 + (lane & 15);
  const int rowBase = blockM + wm * 32 + ((lane >> 4) << 3);
#pragma unroll
  for (int mi = 0; mi < 2; ++mi) {
#pragma unroll
    for (int ni = 0; ni < 4; ++ni) {
      const int ccol = colBase + ni * 16;
      float bv = 0.f;
      if (FUSE_SIGMOID) bv = (ccol < nbias) ? bias[ccol] : 0.f;
#pragma unroll
      for (int v = 0; v < 8; ++v) {
        const int crow = rowBase + mi * 16 + v;
        float val = acc[mi][ni][v];
        if (FUSE_SIGMOID) {
          val = 1.0f / (1.0f + __expf(-(val + bv)));
          ((_Float16*)Cout)[(size_t)crow * ldc + ccol] = (_Float16)val;
        } else {
          ((float*)Cout)[(size_t)crow * ldc + ccol] = val;
        }
      }
    }
  }
}

// ----- Tree expansion: probs[b][L] = prod_d (bit ? g : 1-g), gates staged in LDS -----
__global__ __launch_bounds__(256) void expand_kernel(
    const _Float16* __restrict__ g, _Float16* __restrict__ probs) {
  __shared__ _Float16 sg[4096];
  const int row = blockIdx.x;
  const _Float16* grow = g + (size_t)row * 4096;
  for (int i = threadIdx.x * 8; i < 4096; i += 256 * 8)
    *(v8h*)(&sg[i]) = *(const v8h*)(&grow[i]);
  __syncthreads();
#pragma unroll
  for (int t = 0; t < 16; ++t) {
    const int L = threadIdx.x + t * 256;
    float p = 1.f;
#pragma unroll
    for (int d = 0; d < 12; ++d) {
      const int pos  = L >> (12 - d);
      const int node = (1 << d) - 1 + pos;        // 0 .. 4094
      const int bit  = (L >> (11 - d)) & 1;
      const float gv = (float)sg[node];
      p *= bit ? gv : (1.f - gv);
    }
    probs[(size_t)row * 4096 + L] = (_Float16)p;
  }
}

extern "C" void kernel_launch(void* const* d_in, const int* in_sizes, int n_in,
                              void* d_out, int out_size, void* d_ws, size_t ws_size,
                              hipStream_t stream) {
  const float* x    = (const float*)d_in[0];   // 4096 x 1024
  const float* W    = (const float*)d_in[1];   // 4095 x 1024
  const float* bias = (const float*)d_in[2];   // 4095
  const float* leaf = (const float*)d_in[3];   // 4096 x 1024
  float* out = (float*)d_out;                  // 4096 x 1024

  char* ws = (char*)d_ws;
  _Float16* x16   = (_Float16*)(ws);
  _Float16* W16   = (_Float16*)(ws + ((size_t)8  << 20));
  _Float16* leafT = (_Float16*)(ws + ((size_t)16 << 20));
  _Float16* g16   = (_Float16*)(ws + ((size_t)24 << 20));
  _Float16* p16   = (_Float16*)(ws + ((size_t)56 << 20));

  const int n_x4 = 4096 * 1024 / 4;
  const int n_W4_valid = 4095 * 1024 / 4;
  const int n_W4_total = 4096 * 1024 / 4;

  cvt_f16_kernel<<<(n_x4 + 255) / 256, 256, 0, stream>>>(x, x16, n_x4, n_x4);
  cvt_f16_kernel<<<(n_W4_total + 255) / 256, 256, 0, stream>>>(W, W16, n_W4_valid, n_W4_total);
  transpose_cvt_kernel<<<dim3(32, 128), dim3(32, 8), 0, stream>>>(leaf, leafT);

  // GEMM1: g16 = sigmoid(x16 * W16^T + b), M=N=4096, K=1024
  wmma_gemm_kernel<true><<<dim3(32, 32), 256, 0, stream>>>(
      x16, W16, bias, (void*)g16, 1024, 4096, 4095);

  expand_kernel<<<4096, 256, 0, stream>>>(g16, p16);

  // GEMM2: out = p16 * leafT^T, M=4096, N=1024, K=4096
  wmma_gemm_kernel<false><<<dim3(8, 32), 256, 0, stream>>>(
      p16, leafT, nullptr, (void*)out, 4096, 1024, 0);
}
